// CopiedSetEncoder_9620726743320
// MI455X (gfx1250) — matrix operand order repaired and checked by reading, not compiled, not run on hardware
//
#include <hip/hip_runtime.h>
#include <hip/hip_bf16.h>
#include <math.h>

// Problem dims (match reference)
#define B_    128
#define F_DIM 1024
#define D_IN  128
#define H1_   512
#define H2_   512
#define E_    256
#define H_    256
#define NSHUF 5
#define NEG_INF_F (-1e30f)

typedef __attribute__((ext_vector_type(16))) _Float16 v16h;
typedef __attribute__((ext_vector_type(8)))  _Float16 v8h;
typedef __attribute__((ext_vector_type(8)))  float    v8f;
typedef __attribute__((ext_vector_type(4)))  int      v4i;

// Detect the gfx1250 async global->LDS builtins (ASYNCcnt path).
#if defined(__has_builtin)
#if __has_builtin(__builtin_amdgcn_global_load_async_to_lds_b128) && \
    __has_builtin(__builtin_amdgcn_s_wait_asynccnt)
#define USE_ASYNC_LDS 1
#endif
#endif
#ifndef USE_ASYNC_LDS
#define USE_ASYNC_LDS 0
#endif

#if USE_ASYNC_LDS
// Per hipcc diagnostic: params are pointers to 16-byte int vectors.
typedef __attribute__((address_space(1))) v4i* as1_b128_t;
typedef __attribute__((address_space(3))) v4i* as3_b128_t;
#endif

// LDS tile geometry: rows padded 64 -> 72 halves (16B-aligned, breaks the
// 32-dword row-stride bank pattern).
#define APAD 72
#define BPAD 72

// ---------------------------------------------------------------------------
// LDS fragment loaders (CDNA5 16x16x32 f16 layouts, ISA 7.12.2)
// ---------------------------------------------------------------------------
__device__ __forceinline__ v16h lds_a_frag(const _Float16* __restrict__ base,
                                           int wave, int kk, int lane) {
  const int row = wave * 16 + (lane & 15);
  const int sel = (lane & 16) ? 8 : 0;
  const _Float16* p = base + row * APAD + kk + sel;
  v8h lo = *(const v8h*)(p);
  v8h hi = *(const v8h*)(p + 16);
  v16h r;
#pragma unroll
  for (int i = 0; i < 8; ++i) { r[i] = lo[i]; r[i + 8] = hi[i]; }
  return r;
}

__device__ __forceinline__ v16h lds_b_frag(const _Float16* __restrict__ base,
                                           int ncol0, int kk, int lane) {
  const int col = ncol0 + (lane & 15);
  const int sel = (lane & 16) ? 8 : 0;
  const _Float16* p = base + col * BPAD + kk + sel;
  v8h lo = *(const v8h*)(p);
  v8h hi = *(const v8h*)(p + 16);
  v16h r;
#pragma unroll
  for (int i = 0; i < 8; ++i) { r[i] = lo[i]; r[i + 8] = hi[i]; }
  return r;
}

// ---------------------------------------------------------------------------
// GEMM: C[M,N] = act(A[M,K] * W[N,K]^T + bias[N])
// Block: 256 threads (8 waves), tile 128(M) x 64(N), K-chunks of 64 halves,
// double-buffered through LDS.
// ---------------------------------------------------------------------------
template <bool RELU, int K, int N>
__global__ __launch_bounds__(256) void gemm_bias_act(
    const _Float16* __restrict__ A, const _Float16* __restrict__ W,
    const float* __restrict__ bias, _Float16* __restrict__ Ch,
    float* __restrict__ Cf, int M) {
  __shared__ _Float16 shA[2][128][APAD];  // 2 x 18KB
  __shared__ _Float16 shB[2][64][BPAD];   // 2 x 9KB

  const int t = threadIdx.x;
  const int lane = t & 31;
  const int wave = t >> 5;
  const int mblk = blockIdx.x * 128;
  const int n0 = blockIdx.y * 64;

  // Per-thread staging slice.
  const int ar = t >> 1;          // A row   0..127
  const int ac = (t & 1) * 32;    // A col   0 or 32 (halves)
  const int br = t >> 2;          // B row   0..63
  const int bc = (t & 3) * 16;    // B col   0,16,32,48 (halves)

  const _Float16* gA = A + (size_t)(mblk + ar) * K + ac;  // 64B slice / chunk
  const _Float16* gW = W + (size_t)(n0 + br) * K + bc;    // 32B slice / chunk

#if USE_ASYNC_LDS
  auto stage_issue = [&](int buf, int k0) {
    as1_b128_t ga = (as1_b128_t)(gA + k0);
    as3_b128_t la = (as3_b128_t)&shA[buf][ar][ac];
    __builtin_amdgcn_global_load_async_to_lds_b128(ga, la, 0, 0);
    __builtin_amdgcn_global_load_async_to_lds_b128(ga, la, 16, 0);
    __builtin_amdgcn_global_load_async_to_lds_b128(ga, la, 32, 0);
    __builtin_amdgcn_global_load_async_to_lds_b128(ga, la, 48, 0);
    as1_b128_t gb = (as1_b128_t)(gW + k0);
    as3_b128_t lb = (as3_b128_t)&shB[buf][br][bc];
    __builtin_amdgcn_global_load_async_to_lds_b128(gb, lb, 0, 0);
    __builtin_amdgcn_global_load_async_to_lds_b128(gb, lb, 16, 0);
  };
#endif

  v8f acc0 = {}, acc1 = {}, acc2 = {}, acc3 = {};

#if USE_ASYNC_LDS
  stage_issue(0, 0);
  __builtin_amdgcn_s_wait_asynccnt(0);
  __syncthreads();
#else
  {
    v8h ra0 = *(const v8h*)(gA + 0);
    v8h ra1 = *(const v8h*)(gA + 8);
    v8h ra2 = *(const v8h*)(gA + 16);
    v8h ra3 = *(const v8h*)(gA + 24);
    v8h rb0 = *(const v8h*)(gW + 0);
    v8h rb1 = *(const v8h*)(gW + 8);
    *(v8h*)&shA[0][ar][ac + 0] = ra0;
    *(v8h*)&shA[0][ar][ac + 8] = ra1;
    *(v8h*)&shA[0][ar][ac + 16] = ra2;
    *(v8h*)&shA[0][ar][ac + 24] = ra3;
    *(v8h*)&shB[0][br][bc + 0] = rb0;
    *(v8h*)&shB[0][br][bc + 8] = rb1;
  }
  __syncthreads();
#endif

  int buf = 0;
  for (int k0 = 0; k0 < K; k0 += 64) {
    const bool has_next = (k0 + 64) < K;

#if USE_ASYNC_LDS
    if (has_next) stage_issue(buf ^ 1, k0 + 64);  // DMA overlaps the WMMAs
#else
    v8h ra0, ra1, ra2, ra3, rb0, rb1;
    if (has_next) {  // issue global loads; consumed (ds_store) after compute
      ra0 = *(const v8h*)(gA + k0 + 64 + 0);
      ra1 = *(const v8h*)(gA + k0 + 64 + 8);
      ra2 = *(const v8h*)(gA + k0 + 64 + 16);
      ra3 = *(const v8h*)(gA + k0 + 64 + 24);
      rb0 = *(const v8h*)(gW + k0 + 64 + 0);
      rb1 = *(const v8h*)(gW + k0 + 64 + 8);
    }
#endif

    const _Float16* aBase = &shA[buf][0][0];
    const _Float16* bBase = &shB[buf][0][0];
#pragma unroll
    for (int kk = 0; kk < 64; kk += 32) {
      v16h a = lds_a_frag(aBase, wave, kk, lane);
      v16h b0 = lds_b_frag(bBase, 0, kk, lane);
      v16h b1 = lds_b_frag(bBase, 16, kk, lane);
      v16h b2 = lds_b_frag(bBase, 32, kk, lane);
      v16h b3 = lds_b_frag(bBase, 48, kk, lane);
      acc0 = __builtin_amdgcn_wmma_f32_16x16x32_f16(false, a, false, b0, (short)0, acc0, false, false);
      acc1 = __builtin_amdgcn_wmma_f32_16x16x32_f16(false, a, false, b1, (short)0, acc1, false, false);
      acc2 = __builtin_amdgcn_wmma_f32_16x16x32_f16(false, a, false, b2, (short)0, acc2, false, false);
      acc3 = __builtin_amdgcn_wmma_f32_16x16x32_f16(false, a, false, b3, (short)0, acc3, false, false);
    }

    if (has_next) {
#if USE_ASYNC_LDS
      __builtin_amdgcn_s_wait_asynccnt(0);
#else
      const int nb = buf ^ 1;
      *(v8h*)&shA[nb][ar][ac + 0] = ra0;
      *(v8h*)&shA[nb][ar][ac + 8] = ra1;
      *(v8h*)&shA[nb][ar][ac + 16] = ra2;
      *(v8h*)&shA[nb][ar][ac + 24] = ra3;
      *(v8h*)&shB[nb][br][bc + 0] = rb0;
      *(v8h*)&shB[nb][br][bc + 8] = rb1;
#endif
    }
    __syncthreads();
    buf ^= 1;
  }

  // Epilogue: C/D layout -> row = m0 + vgpr + 8*(lane>=16), col = n0 + (lane&15)
  const int rowBase = mblk + wave * 16 + ((lane & 16) ? 8 : 0);
  const int colOff = lane & 15;
  v8f accs[4] = {acc0, acc1, acc2, acc3};
#pragma unroll
  for (int tN = 0; tN < 4; ++tN) {
    const int col = n0 + tN * 16 + colOff;
    const float bv = bias[col];
#pragma unroll
    for (int r = 0; r < 8; ++r) {
      float v = accs[tN][r] + bv;
      if (RELU) v = fmaxf(v, 0.0f);
      const size_t idx = (size_t)(rowBase + r) * N + col;
      if (Ch) Ch[idx] = (_Float16)v;
      if (Cf) Cf[idx] = v;
    }
  }
}

// ---------------------------------------------------------------------------
// f32 -> f16 conversion (grid-stride)
// ---------------------------------------------------------------------------
__global__ void cvt_f32_f16(const float* __restrict__ in,
                            _Float16* __restrict__ out, size_t n) {
  size_t i = (size_t)blockIdx.x * blockDim.x + threadIdx.x;
  const size_t stride = (size_t)gridDim.x * blockDim.x;
  for (; i < n; i += stride) out[i] = (_Float16)in[i];
}

// ---------------------------------------------------------------------------
// One attention + LSTM step. block = batch element, 256 threads.
// ---------------------------------------------------------------------------
__global__ __launch_bounds__(256) void attn_lstm_step(
    const float* __restrict__ emb, const int* __restrict__ length,
    const float* __restrict__ W_ih, const float* __restrict__ W_hh,
    const float* __restrict__ b_ih, const float* __restrict__ b_hh,
    float* __restrict__ qt, float* __restrict__ ct,
    float* __restrict__ out, int write_out) {
  __shared__ float sh_q[H_];
  __shared__ float sh_w[F_DIM];
  __shared__ float sh_att[E_];
  __shared__ float red[256];

  const int t = threadIdx.x;  // 0..255
  const int b = blockIdx.x;   // 0..127
  const float* embB = emb + (size_t)b * F_DIM * E_;

  sh_q[t] = qt[b * H_ + t];
  __syncthreads();

  const int len = length[b];

  // logits = emb . q, masked
  float lg[4];
  float lmax = -3.4e38f;
#pragma unroll
  for (int j = 0; j < 4; ++j) {
    const int l = t + j * 256;
    const float4* row = (const float4*)(embB + (size_t)l * E_);
    float acc = 0.0f;
    for (int e4 = 0; e4 < E_ / 4; ++e4) {
      const float4 v = row[e4];
      acc += v.x * sh_q[e4 * 4 + 0] + v.y * sh_q[e4 * 4 + 1] +
             v.z * sh_q[e4 * 4 + 2] + v.w * sh_q[e4 * 4 + 3];
    }
    if (l >= len) acc = NEG_INF_F;
    lg[j] = acc;
    lmax = fmaxf(lmax, acc);
  }

  // softmax
  red[t] = lmax;
  __syncthreads();
  for (int s = 128; s > 0; s >>= 1) {
    if (t < s) red[t] = fmaxf(red[t], red[t + s]);
    __syncthreads();
  }
  const float gmax = red[0];
  __syncthreads();

  float lsum = 0.0f;
#pragma unroll
  for (int j = 0; j < 4; ++j) {
    const float w = __expf(lg[j] - gmax);
    sh_w[t + j * 256] = w;
    lsum += w;
  }
  red[t] = lsum;
  __syncthreads();
  for (int s = 128; s > 0; s >>= 1) {
    if (t < s) red[t] += red[t + s];
    __syncthreads();
  }
  const float inv = 1.0f / red[0];
  __syncthreads();

  // attended[e] = sum_l w[l] * emb[l,e]  (thread t == e)
  float att = 0.0f;
  for (int l = 0; l < F_DIM; ++l) {
    att += sh_w[l] * embB[(size_t)l * E_ + t];
  }
  att *= inv;
  sh_att[t] = att;
  __syncthreads();

  // LSTM: thread t computes gates (i,f,g,o) at hidden index t
  float g4[4];
#pragma unroll
  for (int j = 0; j < 4; ++j) {
    const int gi = t + j * H_;
    const float4* wi = (const float4*)(W_ih + (size_t)gi * E_);
    const float4* wh = (const float4*)(W_hh + (size_t)gi * H_);
    float acc = b_ih[gi] + b_hh[gi];
    for (int e4 = 0; e4 < E_ / 4; ++e4) {
      const float4 a = wi[e4];
      acc += a.x * sh_att[e4 * 4 + 0] + a.y * sh_att[e4 * 4 + 1] +
             a.z * sh_att[e4 * 4 + 2] + a.w * sh_att[e4 * 4 + 3];
      const float4 h = wh[e4];
      acc += h.x * sh_q[e4 * 4 + 0] + h.y * sh_q[e4 * 4 + 1] +
             h.z * sh_q[e4 * 4 + 2] + h.w * sh_q[e4 * 4 + 3];
    }
    g4[j] = acc;
  }
  const float ig = 1.0f / (1.0f + __expf(-g4[0]));
  const float fg = 1.0f / (1.0f + __expf(-g4[1]));
  const float gg = tanhf(g4[2]);
  const float og = 1.0f / (1.0f + __expf(-g4[3]));
  const float cn = fg * ct[b * H_ + t] + ig * gg;
  const float hn = og * tanhf(cn);
  ct[b * H_ + t] = cn;
  qt[b * H_ + t] = hn;

  if (write_out) {
    out[(size_t)b * (E_ + H_) + t] = att;
    out[(size_t)b * (E_ + H_) + E_ + t] = hn;
  }
}

// ---------------------------------------------------------------------------
// Launch
// ---------------------------------------------------------------------------
extern "C" void kernel_launch(void* const* d_in, const int* in_sizes, int n_in,
                              void* d_out, int out_size, void* d_ws, size_t ws_size,
                              hipStream_t stream) {
  const float* state = (const float*)d_in[0];
  const int*   length = (const int*)d_in[1];
  const float* W1 = (const float*)d_in[2];
  const float* b1 = (const float*)d_in[3];
  const float* W2 = (const float*)d_in[4];
  const float* b2 = (const float*)d_in[5];
  const float* W3 = (const float*)d_in[6];
  const float* b3 = (const float*)d_in[7];
  const float* W_ih = (const float*)d_in[8];
  const float* W_hh = (const float*)d_in[9];
  const float* b_ih = (const float*)d_in[10];
  const float* b_hh = (const float*)d_in[11];

  char* ws = (char*)d_ws;
  size_t off = 0;
  auto salloc = [&](size_t bytes) -> void* {
    void* p = ws + off;
    off += (bytes + 255) & ~(size_t)255;
    return p;
  };

  const int M = B_ * F_DIM;  // 131072 tokens
  _Float16* Xh  = (_Float16*)salloc((size_t)M * D_IN * 2);
  _Float16* W1h = (_Float16*)salloc((size_t)H1_ * D_IN * 2);
  _Float16* W2h = (_Float16*)salloc((size_t)H2_ * H1_ * 2);
  _Float16* W3h = (_Float16*)salloc((size_t)E_ * H2_ * 2);
  _Float16* H1h = (_Float16*)salloc((size_t)M * H1_ * 2);
  _Float16* H2h = (_Float16*)salloc((size_t)M * H2_ * 2);
  float*    embF = (float*)salloc((size_t)M * E_ * 4);
  float*    qt   = (float*)salloc((size_t)B_ * H_ * 4);
  float*    ct   = (float*)salloc((size_t)B_ * H_ * 4);

  (void)hipMemsetAsync(qt, 0, (size_t)B_ * H_ * 4, stream);
  (void)hipMemsetAsync(ct, 0, (size_t)B_ * H_ * 4, stream);

  // f32 -> f16 conversions
  cvt_f32_f16<<<dim3(2048), dim3(256), 0, stream>>>(state, Xh, (size_t)M * D_IN);
  cvt_f32_f16<<<dim3(256), dim3(256), 0, stream>>>(W1, W1h, (size_t)H1_ * D_IN);
  cvt_f32_f16<<<dim3(512), dim3(256), 0, stream>>>(W2, W2h, (size_t)H2_ * H1_);
  cvt_f32_f16<<<dim3(512), dim3(256), 0, stream>>>(W3, W3h, (size_t)E_ * H2_);

  // MLP: Linear+ReLU, Linear+ReLU, Linear
  gemm_bias_act<true, D_IN, H1_><<<dim3(M / 128, H1_ / 64), dim3(256), 0, stream>>>(
      Xh, W1h, b1, H1h, nullptr, M);
  gemm_bias_act<true, H1_, H2_><<<dim3(M / 128, H2_ / 64), dim3(256), 0, stream>>>(
      H1h, W2h, b2, H2h, nullptr, M);
  gemm_bias_act<false, H2_, E_><<<dim3(M / 128, E_ / 64), dim3(256), 0, stream>>>(
      H2h, W3h, b3, nullptr, embF, M);

  // 5 sequential attention + LSTM steps
  for (int s = 0; s < NSHUF; ++s) {
    attn_lstm_step<<<dim3(B_), dim3(256), 0, stream>>>(
        embF, length, W_ih, W_hh, b_ih, b_hh, qt, ct,
        (float*)d_out, (s == NSHUF - 1) ? 1 : 0);
  }
}
